// Model_876173328517
// MI455X (gfx1250) — matrix operand order repaired and checked by reading, not compiled, not run on hardware
//
#include <hip/hip_runtime.h>
#include <hip/hip_bf16.h>

typedef __attribute__((ext_vector_type(16))) _Float16 v16h;
typedef __attribute__((ext_vector_type(8)))  float    v8f;

union Frag16 { v16h v; _Float16 h[16]; uint4 q[2]; };

#define WMMA_F16(a, b, c) \
  __builtin_amdgcn_wmma_f32_16x16x32_f16(false, (a), false, (b), (short)0, (c), false, false)

// Wave-private LDS producer->consumer sync: drain DS counter, then a
// scheduling barrier so the compiler can't hoist the consumer loads.
__device__ __forceinline__ void wave_lds_sync() {
  asm volatile("s_wait_dscnt 0" ::: "memory");
  __builtin_amdgcn_wave_barrier();
}

// Same, but also drains the async (memory->LDS) engine.
__device__ __forceinline__ void wave_async_lds_sync() {
  asm volatile("s_wait_asynccnt 0" ::: "memory");
  asm volatile("s_wait_dscnt 0" ::: "memory");
  __builtin_amdgcn_wave_barrier();
}

// CDNA5 async copy: 16 bytes memory -> LDS, per enabled lane. GV addressing.
__device__ __forceinline__ void async_copy_b128(uint32_t lds_byte_off, const void* gaddr) {
  asm volatile("global_load_async_to_lds_b128 %0, %1, off"
               :: "v"(lds_byte_off), "v"(gaddr) : "memory");
}

// LDS byte offset of a generic (shared) pointer: low 32 bits of the flat addr.
__device__ __forceinline__ uint32_t lds_off(const void* p) {
  return (uint32_t)(uintptr_t)p;
}

// A-fragment (16x32 f16) from an LDS row-major [16][stride] tile, K-chunk k.
// ISA layout: lane<16 -> elems0..7:K=0..7, elems8..15:K=16..23 ; lane>=16 -> +8.
__device__ __forceinline__ v16h load_a_frag(const _Float16* t, int stride, int k, int lane) {
  int row = lane & 15;
  int b = (lane < 16) ? 0 : 8;
  const _Float16* p = t + row * stride + k * 32;
  Frag16 f;
  f.q[0] = *(const uint4*)(p + b);
  f.q[1] = *(const uint4*)(p + 16 + b);
  return f.v;
}

// B-fragment (32x16 f16) pre-swizzled in global memory: frag*512 + lane*16 halves.
__device__ __forceinline__ v16h load_b_frag(const _Float16* w, int frag, int lane) {
  const _Float16* p = w + (size_t)frag * 512 + (size_t)lane * 16;
  Frag16 f;
  f.q[0] = *(const uint4*)(p);
  f.q[1] = *(const uint4*)(p + 8);
  return f.v;
}

// ---------------------------------------------------------------------------
// Convert an f32 row-major weight [F][Nsrc] into f16 B-fragments.
// Element (frag=k*Ntiles+nt, lane, e) <- W[k*32 + (lane<16?0:8) + (e<8?e:e+8)][nt*16 + lane%16]
__global__ void k_swizzle_w(const float* __restrict__ W, _Float16* __restrict__ dst,
                            int F, int Nsrc, int Ntiles, int total) {
  int i = blockIdx.x * blockDim.x + threadIdx.x;
  if (i >= total) return;
  int e    = i & 15;
  int lane = (i >> 4) & 31;
  int frag = i >> 9;
  int nt = frag % Ntiles;
  int k  = frag / Ntiles;
  int K = k * 32 + ((lane < 16) ? 0 : 8) + ((e < 8) ? e : e + 8);
  int N = nt * 16 + (lane & 15);
  float val = (K < F && N < Nsrc) ? W[(size_t)K * Nsrc + N] : 0.f;
  dst[i] = (_Float16)val;
}

__global__ void k_zero(float* __restrict__ p, int n) {
  int i = blockIdx.x * blockDim.x + threadIdx.x;
  if (i < n) p[i] = 0.f;
}

// BN training mode: scale = gamma*rsqrt(var+eps); shift = beta - mean*scale.
__global__ void k_bn_finalize(const float* __restrict__ acc, const float* __restrict__ gamma,
                              const float* __restrict__ beta, float invN, float* __restrict__ ss) {
  int c = threadIdx.x;
  if (c < 64) {
    float m   = acc[c] * invN;
    float var = acc[64 + c] * invN - m * m;
    float s   = gamma[c] * rsqrtf(var + 1e-5f);
    ss[c]      = s;
    ss[64 + c] = beta[c] - m * s;
  }
}

// Apply BN affine in-place; optionally emit an f16 row-major copy that the
// edge kernels gather with the async-LDS path.
__global__ void k_bn_apply(float* __restrict__ Y, _Float16* __restrict__ Yh,
                           const float* __restrict__ ss, int n) {
  int i = blockIdx.x * blockDim.x + threadIdx.x;
  if (i < n) {
    int c = i & 63;
    float y = Y[i] * ss[c] + ss[64 + c];
    Y[i] = y;
    if (Yh) Yh[i] = (_Float16)y;
  }
}

// ---------------------------------------------------------------------------
// Edge MLP g: gin=[U[idxU],Vv[idxV],ev] (129 -> pad 160) ->64 relu ->64 relu,
// then atomic segment-sum scatter into agg[idxU]. One wave per 16 edges.
// U/Vv are pre-converted f16 rows (64 halves = 128B each); the gather is a
// pure memory->LDS DMA via global_load_async_to_lds_b128.
__global__ __launch_bounds__(128) void k_edge_mlp(
    const _Float16* __restrict__ U, const _Float16* __restrict__ Vv,
    const int* __restrict__ idxU, const int* __restrict__ idxV,
    const float* __restrict__ ev,
    const _Float16* __restrict__ w1, const float* __restrict__ b1,
    const _Float16* __restrict__ w2, const float* __restrict__ b2,
    float* __restrict__ agg, int nE) {
  __shared__ alignas(16) _Float16 sA[4][16 * 160];
  __shared__ alignas(16) _Float16 sH[4][16 * 64];
  int wv = threadIdx.x >> 5, lane = threadIdx.x & 31;
  int base = (blockIdx.x * 4 + wv) * 16;
  _Float16* At = sA[wv];
  _Float16* Ht = sH[wv];
  uint32_t atBase = lds_off(At);

  // Async gather: 8 lanes move one 128B f16 row; 32 lanes = 4 rows per issue.
  int chunk = lane & 7;   // 16B chunk within a 128B row segment
  int rg    = lane >> 3;  // row within group of 4
#pragma unroll
  for (int i = 0; i < 4; ++i) {
    int r = i * 4 + rg;
    int e = base + r;
    bool ok = e < nE;
    int iu = ok ? idxU[e] : 0;
    int iv = ok ? idxV[e] : 0;
    const char* gu = (const char*)(U  + (size_t)iu * 64) + chunk * 16;
    const char* gv = (const char*)(Vv + (size_t)iv * 64) + chunk * 16;
    uint32_t du = atBase + (uint32_t)(r * 320 + chunk * 16); // cols 0..63
    async_copy_b128(du, gu);
    async_copy_b128(du + 128, gv);                           // cols 64..127
  }
  // Tail cols 128..159: [eval, 0, 0, ...]; one lane per row.
  if (lane < 16) {
    int e = base + lane;
    float evv = (e < nE) ? ev[e] : 0.f;
    _Float16* row = At + lane * 160 + 128;
    uint4 z = {0u, 0u, 0u, 0u};
    *(uint4*)(row)      = z;
    *(uint4*)(row + 8)  = z;
    *(uint4*)(row + 16) = z;
    *(uint4*)(row + 24) = z;
    row[0] = (_Float16)evv;
  }
  wave_async_lds_sync();

  // Layer 1: 16x160 @ 160x64 via 5 K-chunks x 4 N-tiles of WMMA.
  v8f acc[4] = {};
#pragma unroll
  for (int k = 0; k < 5; ++k) {
    v16h a = load_a_frag(At, 160, k, lane);
#pragma unroll
    for (int nt = 0; nt < 4; ++nt)
      acc[nt] = WMMA_F16(a, load_b_frag(w1, k * 4 + nt, lane), acc[nt]);
  }
  int laneN = lane & 15, rbase = (lane < 16) ? 0 : 8;
#pragma unroll
  for (int nt = 0; nt < 4; ++nt)
#pragma unroll
    for (int r = 0; r < 8; ++r) {
      float h = acc[nt][r] + b1[nt * 16 + laneN];
      h = h > 0.f ? h : 0.f;
      Ht[(rbase + r) * 64 + nt * 16 + laneN] = (_Float16)h;
    }
  wave_lds_sync();

  // Layer 2: 16x64 @ 64x64.
  v8f acc2[4] = {};
#pragma unroll
  for (int k = 0; k < 2; ++k) {
    v16h a = load_a_frag(Ht, 64, k, lane);
#pragma unroll
    for (int nt = 0; nt < 4; ++nt)
      acc2[nt] = WMMA_F16(a, load_b_frag(w2, k * 4 + nt, lane), acc2[nt]);
  }
  // ReLU + atomic scatter segment-sum.
#pragma unroll
  for (int r = 0; r < 8; ++r) {
    int e = base + rbase + r;
    if (e < nE) {
      int iu = idxU[e];
#pragma unroll
      for (int nt = 0; nt < 4; ++nt) {
        int col = nt * 16 + laneN;
        float y = acc2[nt][r] + b2[col];
        y = y > 0.f ? y : 0.f;
        atomicAdd(&agg[(size_t)iu * 64 + col], y);
      }
    }
  }
}

// ---------------------------------------------------------------------------
// Node MLP: X = concat(X1[F1], X2[F2]) -> 64 relu -> NOUT (+relu / +sigmoid),
// optional fused BN sum / sum^2 accumulation. One wave per 16 rows.
template <int NOUT, bool RELU2, bool SIGMOID, bool BNACC>
__global__ __launch_bounds__(128) void k_node_mlp(
    const float* __restrict__ X1, int F1, const float* __restrict__ X2, int F2,
    const _Float16* __restrict__ w1, const float* __restrict__ b1,
    const _Float16* __restrict__ w2, const float* __restrict__ b2,
    float* __restrict__ Y, int nRows, float* __restrict__ bnAcc) {
  constexpr int NT2 = (NOUT + 15) / 16;
  __shared__ alignas(16) _Float16 sA[4][16 * 128];
  __shared__ alignas(16) _Float16 sH[4][16 * 64];
  int wv = threadIdx.x >> 5, lane = threadIdx.x & 31;
  int base = (blockIdx.x * 4 + wv) * 16;
  _Float16* At = sA[wv];
  _Float16* Ht = sH[wv];
  int F = F1 + F2;
  int Kp = (F + 31) & ~31;

  for (int r = 0; r < 16; ++r) {
    int row = base + r;
    bool ok = row < nRows;
    for (int cc = lane; cc < Kp; cc += 32) {
      float val = 0.f;
      if (ok) {
        if (cc < F1)     val = X1[(size_t)row * F1 + cc];
        else if (cc < F) val = X2[(size_t)row * F2 + (cc - F1)];
      }
      At[r * 128 + cc] = (_Float16)val;
    }
  }
  wave_lds_sync();

  v8f acc[4] = {};
  int nk = Kp >> 5;
  for (int k = 0; k < nk; ++k) {
    v16h a = load_a_frag(At, 128, k, lane);
#pragma unroll
    for (int nt = 0; nt < 4; ++nt)
      acc[nt] = WMMA_F16(a, load_b_frag(w1, k * 4 + nt, lane), acc[nt]);
  }
  int laneN = lane & 15, rbase = (lane < 16) ? 0 : 8;
#pragma unroll
  for (int nt = 0; nt < 4; ++nt)
#pragma unroll
    for (int r = 0; r < 8; ++r) {
      float h = acc[nt][r] + b1[nt * 16 + laneN];
      h = h > 0.f ? h : 0.f;
      Ht[(rbase + r) * 64 + nt * 16 + laneN] = (_Float16)h;
    }
  wave_lds_sync();

  v8f acc2[NT2] = {};
#pragma unroll
  for (int k = 0; k < 2; ++k) {
    v16h a = load_a_frag(Ht, 64, k, lane);
#pragma unroll
    for (int nt = 0; nt < NT2; ++nt)
      acc2[nt] = WMMA_F16(a, load_b_frag(w2, k * NT2 + nt, lane), acc2[nt]);
  }
#pragma unroll
  for (int nt = 0; nt < NT2; ++nt)
#pragma unroll
    for (int r = 0; r < 8; ++r) {
      int row = base + rbase + r;
      int col = nt * 16 + laneN;
      float y = acc2[nt][r] + ((col < NOUT) ? b2[col] : 0.f);
      if (RELU2)   y = y > 0.f ? y : 0.f;
      if (SIGMOID) y = 1.f / (1.f + __expf(-y));
      if (row < nRows && col < NOUT) {
        Y[(size_t)row * NOUT + col] = y;
        if (BNACC) {
          atomicAdd(&bnAcc[col], y);
          atomicAdd(&bnAcc[64 + col], y * y);
        }
      }
    }
}

// ---------------------------------------------------------------------------
extern "C" void kernel_launch(void* const* d_in, const int* in_sizes, int n_in,
                              void* d_out, int out_size, void* d_ws, size_t ws_size,
                              hipStream_t stream) {
  const int NV = 50000, NC = 50000, NE = 1200000, Dm = 64;
  const float* v    = (const float*)d_in[0];  // (NV,19)
  const float* cF   = (const float*)d_in[1];  // (NC,5)
  const int*   erow = (const int*)d_in[2];    // (NE,) constraint idx
  const int*   ecol = (const int*)d_in[3];    // (NE,) variable idx
  const float* ev   = (const float*)d_in[4];  // (NE,1)
  // params flattened in dict insertion order, tuples in order:
  // 0..3 emb_v(W1,b1,W2,b2) 4,5 bn_v(g,b) 6..9 emb_c 10,11 bn_c
  // 12..15 c_g 16..19 c_f 20,21 bn_c2 22..25 v_g 26..29 v_f 30,31 bn_v2 32..35 tail
  const float* P[36];
  for (int i = 0; i < 36; ++i) P[i] = (const float*)d_in[5 + i];

  float* f = (float*)d_ws;
  float* v1  = f; f += (size_t)NV * Dm;
  float* c1  = f; f += (size_t)NC * Dm;
  float* c2  = f; f += (size_t)NC * Dm;
  float* v2  = f; f += (size_t)NV * Dm;
  float* agg = f; f += (size_t)NV * Dm;
  float* bnA = f; f += 128;   // [sum(64) | sumsq(64)]
  float* ss  = f; f += 128;   // [scale(64) | shift(64)]
  _Float16* wh = (_Float16*)f;

  // Pre-swizzle all weight matrices into f16 B-fragments.
  struct WD { const float* W; int F, Nsrc, Nt; };
  WD wd[14] = {
    {P[0], 19, 64, 4}, {P[2], 64, 64, 4},    // emb_v W1,W2
    {P[6], 5, 64, 4},  {P[8], 64, 64, 4},    // emb_c
    {P[12], 129, 64, 4}, {P[14], 64, 64, 4}, // c_g
    {P[16], 128, 64, 4}, {P[18], 64, 64, 4}, // c_f
    {P[22], 129, 64, 4}, {P[24], 64, 64, 4}, // v_g
    {P[26], 128, 64, 4}, {P[28], 64, 64, 4}, // v_f
    {P[32], 64, 64, 4},  {P[34], 64, 8, 1},  // tail W1, W2(64x8 padded to 16 cols)
  };
  _Float16* wp[14];
  size_t off = 0;
  for (int i = 0; i < 14; ++i) {
    int Kp = (wd[i].F + 31) & ~31;
    int total = Kp * wd[i].Nt * 16;
    wp[i] = wh + off;
    off += (size_t)total;
    k_swizzle_w<<<(total + 255) / 256, 256, 0, stream>>>(wd[i].W, wp[i], wd[i].F,
                                                         wd[i].Nsrc, wd[i].Nt, total);
  }
  // f16 row-major node-feature copies for the async edge gather.
  _Float16* v1h = wh + off; off += (size_t)NV * Dm;
  _Float16* c1h = wh + off; off += (size_t)NC * Dm;
  _Float16* c2h = wh + off; off += (size_t)NC * Dm;

  dim3 blk(128);
  int nodeTiles = (NV + 15) / 16;  // NV == NC
  dim3 nodeGrid((nodeTiles + 3) / 4);
  int nElem = NV * Dm;
  int elemBlocks = (nElem + 255) / 256;
  int edgeBlocks = ((NE + 15) / 16 + 3) / 4;

  // v1 = BN(mlp2(v, emb_v))  (+ f16 copy)
  k_zero<<<1, 256, 0, stream>>>(bnA, 128);
  k_node_mlp<64, true, false, true><<<nodeGrid, blk, 0, stream>>>(
      v, 19, nullptr, 0, wp[0], P[1], wp[1], P[3], v1, NV, bnA);
  k_bn_finalize<<<1, 64, 0, stream>>>(bnA, P[4], P[5], 1.f / NV, ss);
  k_bn_apply<<<elemBlocks, 256, 0, stream>>>(v1, v1h, ss, nElem);

  // c1 = BN(mlp2(c, emb_c))  (+ f16 copy)
  k_zero<<<1, 256, 0, stream>>>(bnA, 128);
  k_node_mlp<64, true, false, true><<<nodeGrid, blk, 0, stream>>>(
      cF, 5, nullptr, 0, wp[2], P[7], wp[3], P[9], c1, NC, bnA);
  k_bn_finalize<<<1, 64, 0, stream>>>(bnA, P[10], P[11], 1.f / NC, ss);
  k_bn_apply<<<elemBlocks, 256, 0, stream>>>(c1, c1h, ss, nElem);

  // c-side half conv: agg over constraints, then c2 = BN(f(concat[c1, agg]))
  k_zero<<<elemBlocks, 256, 0, stream>>>(agg, nElem);
  k_edge_mlp<<<edgeBlocks, blk, 0, stream>>>(c1h, v1h, erow, ecol, ev,
                                             wp[4], P[13], wp[5], P[15], agg, NE);
  k_zero<<<1, 256, 0, stream>>>(bnA, 128);
  k_node_mlp<64, true, false, true><<<nodeGrid, blk, 0, stream>>>(
      c1, 64, agg, 64, wp[6], P[17], wp[7], P[19], c2, NC, bnA);
  k_bn_finalize<<<1, 64, 0, stream>>>(bnA, P[20], P[21], 1.f / NC, ss);
  k_bn_apply<<<elemBlocks, 256, 0, stream>>>(c2, c2h, ss, nElem);

  // v-side half conv: agg over variables, then v2 = BN(f(concat[v1, agg]))
  k_zero<<<elemBlocks, 256, 0, stream>>>(agg, nElem);
  k_edge_mlp<<<edgeBlocks, blk, 0, stream>>>(v1h, c2h, ecol, erow, ev,
                                             wp[8], P[23], wp[9], P[25], agg, NE);
  k_zero<<<1, 256, 0, stream>>>(bnA, 128);
  k_node_mlp<64, true, false, true><<<nodeGrid, blk, 0, stream>>>(
      v1, 64, agg, 64, wp[10], P[27], wp[11], P[29], v2, NV, bnA);
  k_bn_finalize<<<1, 64, 0, stream>>>(bnA, P[30], P[31], 1.f / NV, ss);
  k_bn_apply<<<elemBlocks, 256, 0, stream>>>(v2, nullptr, ss, nElem);

  // tail: sigmoid(mlp2(v2, tail, relu_out=False)) -> (NV, 8) float32
  k_node_mlp<8, false, true, false><<<nodeGrid, blk, 0, stream>>>(
      v2, 64, nullptr, 0, wp[12], P[33], wp[13], P[35], (float*)d_out, NV, nullptr);
}
